// Network_60309930770665
// MI455X (gfx1250) — matrix operand order repaired and checked by reading, not compiled, not run on hardware
//
#include <hip/hip_runtime.h>
#include <hip/hip_bf16.h>
#include <math.h>

typedef __attribute__((ext_vector_type(16))) _Float16 v16h;
typedef __attribute__((ext_vector_type(8)))  _Float16 v8h;
typedef __attribute__((ext_vector_type(8)))  float    v8f;

#define NRAYS 2048
#define NSC   64           // coarse samples per ray
#define NSF   128          // fine samples per ray (64 + 64 importance)
#define MC    (NRAYS*NSC)  // 131072 coarse rows
#define MF    (NRAYS*NSF)  // 262144 fine rows

// ---------------------------------------------------------------------------
// Weight pre-pack: f32 (K,N) row-major -> f16, zero-padded to (Kpad,Npad),
// swizzled into this file's B layout for v_wmma_f32_16x16x32_f16 (contiguous k, same as A):
//   frag(kt,nt): lane = half*16+n ; u32 v holds B[k0+2v], B[k0+2v+1] (col n)
//   where k0 = kt*32 + half*16.  Each lane's 16 halfs are contiguous.
// ---------------------------------------------------------------------------
__global__ void pack_weight(const float* __restrict__ W, _Float16* __restrict__ dst,
                            int K, int N, int Kpad, int Npad)
{
    int idx = blockIdx.x * blockDim.x + threadIdx.x;    // (kt*NtTot + nt)*32 + lane
    int total = (Kpad / 32) * (Npad / 16) * 32;
    if (idx >= total) return;
    int lane = idx & 31;
    int t    = idx >> 5;
    int ntTot = Npad / 16;
    int nt = t % ntTot;
    int kt = t / ntTot;
    int half = lane >> 4;
    int n    = lane & 15;
    int col  = nt * 16 + n;
    _Float16* d = dst + (size_t)idx * 16;
#pragma unroll
    for (int v = 0; v < 8; ++v) {
        int k0 = kt * 32 + half * 16 + 2 * v;
        float a0 = (k0     < K && col < N) ? W[(size_t)k0       * N + col] : 0.f;
        float a1 = (k0 + 1 < K && col < N) ? W[(size_t)(k0 + 1) * N + col] : 0.f;
        d[2 * v]     = (_Float16)a0;
        d[2 * v + 1] = (_Float16)a1;
    }
}

// ---------------------------------------------------------------------------
// WMMA GEMM:  Out = epilogue( A[M x KT*32] * Wp[KT*32 x NTTOT*16] + bias )
// block = 128 threads (4 waves), each wave -> 16 rows x (NT*16) cols.
// All shape parameters are compile-time: the K loop is FULLY UNROLLED with a
// 2-deep software pipeline (loads for kt+1 issued before WMMAs of kt).  After
// unrolling, buffer rotation is pure SSA renaming (no v_mov) and every load
// address is base + compile-time immediate offset -> no address math and
// partial s_wait_loadcnt values in the hot code.
// MODE 0: A from single f16 matrix (row stride LDA)
// MODE 1: layer-5 concat: K tiles 0..7 from A (LDA), tile>=8 from A2 (LDA2)
// MODE 2: views concat: K tiles 0..7 from A, tile 8 from per-ray A2[m/S]
// EPI  0: relu -> f16 (LDO)    1: linear -> f16 (LDO)
// EPI  2: sigma -> raw[m*4+3]  3: rgb -> raw[m*4 + n] (n<3)
// ---------------------------------------------------------------------------
template<int MODE, int NT, int EPI, int KT, int NTTOT, int LDA, int LDA2, int LDO>
__global__ __launch_bounds__(128)
void gemm_wmma(const _Float16* __restrict__ A, const _Float16* __restrict__ A2,
               const _Float16* __restrict__ Wp, const float* __restrict__ bias,
               void* __restrict__ Out, int S)
{
    const int lane  = threadIdx.x & 31;
    const int wave  = threadIdx.x >> 5;
    const int mBase = blockIdx.y * 64 + wave * 16;
    const int ntBase = blockIdx.x * NT;
    const int half = lane >> 4;
    const int l15  = lane & 15;

    const _Float16* wlane = Wp + ((size_t)ntBase * 32 + lane) * 16;
    const _Float16* ar    = A + (size_t)(mBase + l15) * LDA + half * 16;
    const _Float16* ar2   = nullptr;
    if (MODE == 1) ar2 = A2 + (size_t)(mBase + l15) * LDA2 + half * 16;
    if (MODE == 2) ar2 = A2 + (size_t)((mBase + l15) / S) * LDA2 + half * 16;

    v8f c[NT];
#pragma unroll
    for (int j = 0; j < NT; ++j)
#pragma unroll
        for (int i = 0; i < 8; ++i) c[j][i] = 0.f;

    // A-fragment source pointer for (compile-time) tile kt
    auto aPtr = [&](int kt) -> const _Float16* {
        return (MODE == 0 || kt < 8) ? (ar + kt * 32) : (ar2 + (kt - 8) * 32);
    };

    // ---- 2-deep pipelined, fully unrolled K loop ----
    v8h  lo_n, hi_n;
    v16h b_n[NT];
    {
        const _Float16* p = aPtr(0);
        lo_n = *(const v8h*)p;
        hi_n = *(const v8h*)(p + 8);
#pragma unroll
        for (int j = 0; j < NT; ++j) b_n[j] = *(const v16h*)(wlane + j * 512);
    }
#pragma unroll
    for (int kt = 0; kt < KT; ++kt) {
        v8h  lo_c = lo_n, hi_c = hi_n;
        v16h b_c[NT];
#pragma unroll
        for (int j = 0; j < NT; ++j) b_c[j] = b_n[j];
        if (kt + 1 < KT) {                       // prefetch next fragment set
            const _Float16* p = aPtr(kt + 1);
            lo_n = *(const v8h*)p;
            hi_n = *(const v8h*)(p + 8);
#pragma unroll
            for (int j = 0; j < NT; ++j)
                b_n[j] = *(const v16h*)(wlane + (size_t)(kt + 1) * (NTTOT * 512) + j * 512);
        }
        v16h a;
#pragma unroll
        for (int i = 0; i < 8; ++i) { a[i] = lo_c[i]; a[8 + i] = hi_c[i]; }
#pragma unroll
        for (int j = 0; j < NT; ++j)
            c[j] = __builtin_amdgcn_wmma_f32_16x16x32_f16(
                       false, a, false, b_c[j], (short)0, c[j], false, false);
    }

    // ---- epilogue ----
#pragma unroll
    for (int j = 0; j < NT; ++j) {
        int n0   = (ntBase + j) * 16;
        int ncol = n0 + l15;
        float bn = 0.f;
        if (EPI <= 1) bn = bias[ncol];
#pragma unroll
        for (int r = 0; r < 8; ++r) {
            int m   = mBase + half * 8 + r;
            float v = c[j][r] + bn;
            if (EPI == 0) {
                ((_Float16*)Out)[(size_t)m * LDO + ncol] = (_Float16)fmaxf(v, 0.f);
            } else if (EPI == 1) {
                ((_Float16*)Out)[(size_t)m * LDO + ncol] = (_Float16)v;
            } else if (EPI == 2) {
                if (ncol == 0) ((float*)Out)[(size_t)m * 4 + 3] = v + bias[0];
            } else {
                if (ncol < 3) ((float*)Out)[(size_t)m * 4 + ncol] = v + bias[ncol];
            }
        }
    }
}

// ---------------------------------------------------------------------------
// Coarse z values: linspace(0,100,64) per ray
// ---------------------------------------------------------------------------
__global__ void gen_z_coarse(float* __restrict__ z)
{
    int m = blockIdx.x * blockDim.x + threadIdx.x;
    if (m >= MC) return;
    z[m] = 100.f * (float)(m & 63) / 63.f;
}

// posenc(xyz, 10) -> X0 row (63 used + 1 zero pad)
__global__ void encode_pts(const float* __restrict__ rays, const float* __restrict__ z,
                           _Float16* __restrict__ X0, int S, int M)
{
    int m = blockIdx.x * blockDim.x + threadIdx.x;
    if (m >= M) return;
    int r = m / S;
    const float* rp = rays + (size_t)r * 6;
    float ox = rp[0], oy = rp[1], oz = rp[2];
    float dx = rp[3], dy = rp[4], dz = rp[5];
    float sc = sqrtf(dx * dx + dy * dy + dz * dz);
    float zz = z[m];
    float px = ox + dx * zz / sc, py = oy + dy * zz / sc, pz = oz + dz * zz / sc;
    _Float16* o = X0 + (size_t)m * 64;
    o[0] = (_Float16)px; o[1] = (_Float16)py; o[2] = (_Float16)pz;
    float f = 1.f;
#pragma unroll
    for (int l = 0; l < 10; ++l) {
        int b = 3 + 6 * l;
        o[b + 0] = (_Float16)sinf(px * f);
        o[b + 1] = (_Float16)sinf(py * f);
        o[b + 2] = (_Float16)sinf(pz * f);
        o[b + 3] = (_Float16)cosf(px * f);
        o[b + 4] = (_Float16)cosf(py * f);
        o[b + 5] = (_Float16)cosf(pz * f);
        f *= 2.f;
    }
    o[63] = (_Float16)0.f;
}

// posenc(dir, 4) -> 27 cols (+5 zero pad), plus per-ray |d|
__global__ void encode_dirs(const float* __restrict__ rays,
                            _Float16* __restrict__ denc, float* __restrict__ scaleBuf)
{
    int r = blockIdx.x * blockDim.x + threadIdx.x;
    if (r >= NRAYS) return;
    const float* rp = rays + (size_t)r * 6;
    float dx = rp[3], dy = rp[4], dz = rp[5];
    scaleBuf[r] = sqrtf(dx * dx + dy * dy + dz * dz);
    _Float16* o = denc + (size_t)r * 32;
    o[0] = (_Float16)dx; o[1] = (_Float16)dy; o[2] = (_Float16)dz;
    float f = 1.f;
#pragma unroll
    for (int l = 0; l < 4; ++l) {
        int b = 3 + 6 * l;
        o[b + 0] = (_Float16)sinf(dx * f);
        o[b + 1] = (_Float16)sinf(dy * f);
        o[b + 2] = (_Float16)sinf(dz * f);
        o[b + 3] = (_Float16)cosf(dx * f);
        o[b + 4] = (_Float16)cosf(dy * f);
        o[b + 5] = (_Float16)cosf(dz * f);
        f *= 2.f;
    }
    for (int c = 27; c < 32; ++c) o[c] = (_Float16)0.f;
}

// raw2outputs: per-ray serial cumprod. raw = [rgb0..2, sigma] per sample.
__global__ void render_rays(const float* __restrict__ raw, const float* __restrict__ z,
                            const float* __restrict__ scale, int S,
                            float* __restrict__ wOut, float* __restrict__ rgbOut,
                            float* __restrict__ depthOut, float* __restrict__ accOut)
{
    int r = blockIdx.x * blockDim.x + threadIdx.x;
    if (r >= NRAYS) return;
    float trans = 1.f, c0 = 0.f, c1 = 0.f, c2 = 0.f, dep = 0.f, acc = 0.f;
    float sc = scale[r];
    for (int s = 0; s < S; ++s) {
        const float* rw = raw + ((size_t)r * S + s) * 4;
        float sigma = fmaxf(rw[3], 0.f);
        float zz = z[(size_t)r * S + s];
        float dist = (s < S - 1) ? (z[(size_t)r * S + s + 1] - zz) : 1e10f * sc;
        float alpha = 1.f - expf(-sigma * dist);
        float w = alpha * trans;
        trans *= (1.f - alpha + 1e-10f);
        if (wOut) wOut[(size_t)r * S + s] = w;
        if (rgbOut) {
            c0 += w * (1.f / (1.f + expf(-rw[0])));
            c1 += w * (1.f / (1.f + expf(-rw[1])));
            c2 += w * (1.f / (1.f + expf(-rw[2])));
            dep += w * zz / sc;
            acc += w;
        }
    }
    if (rgbOut) {
        rgbOut[r * 3 + 0] = c0; rgbOut[r * 3 + 1] = c1; rgbOut[r * 3 + 2] = c2;
        depthOut[r] = dep; accOut[r] = acc;
    }
}

__device__ __forceinline__ float rnd01(unsigned x)
{
    x ^= x >> 17; x *= 0xed5ad4bbu;
    x ^= x >> 11; x *= 0xac4c1b51u;
    x ^= x >> 15; x *= 0x31848babu;
    x ^= x >> 14;
    return (float)(x >> 8) * (1.0f / 16777216.0f);
}

// inverse-CDF importance sampling over 63 mid-bins (weights0[...,1:-1])
__global__ void sample_pdf_k(const float* __restrict__ w0, const float* __restrict__ zc,
                             float* __restrict__ zs)
{
    int r = blockIdx.x * blockDim.x + threadIdx.x;
    if (r >= NRAYS) return;
    float wv[62]; float sum = 0.f;
    for (int i = 0; i < 62; ++i) { wv[i] = w0[(size_t)r * 64 + 1 + i] + 1e-5f; sum += wv[i]; }
    float cdf[63]; cdf[0] = 0.f; float run = 0.f;
    for (int i = 0; i < 62; ++i) { run += wv[i] / sum; cdf[i + 1] = run; }
    float zm[63];
    for (int j = 0; j < 63; ++j)
        zm[j] = 0.5f * (zc[(size_t)r * 64 + j] + zc[(size_t)r * 64 + j + 1]);
    for (int j = 0; j < 64; ++j) {
        float u = rnd01((unsigned)(r * 64 + j) * 2654435761u + 12345u);
        int ind = 0;
        for (int i = 0; i < 63; ++i) ind += (cdf[i] <= u) ? 1 : 0;   // searchsorted right
        int below = ind - 1; if (below < 0) below = 0; if (below > 62) below = 62;
        int above = ind;     if (above > 62) above = 62;
        float denom = cdf[above] - cdf[below];
        if (denom < 1e-5f) denom = 1.f;
        float t = (u - cdf[below]) / denom;
        zs[(size_t)r * 64 + j] = zm[below] + t * (zm[above] - zm[below]);
    }
}

// merge coarse z (sorted) with importance samples and sort all 128
__global__ void sort_z(const float* __restrict__ zc, const float* __restrict__ zs,
                       float* __restrict__ zall)
{
    int r = blockIdx.x * blockDim.x + threadIdx.x;
    if (r >= NRAYS) return;
    float a[128];
    for (int i = 0; i < 64; ++i) { a[i] = zc[(size_t)r * 64 + i]; a[64 + i] = zs[(size_t)r * 64 + i]; }
    for (int i = 1; i < 128; ++i) {
        float key = a[i]; int j = i - 1;
        while (j >= 0 && a[j] > key) { a[j + 1] = a[j]; --j; }
        a[j + 1] = key;
    }
    for (int i = 0; i < 128; ++i) zall[(size_t)r * 128 + i] = a[i];
}

// ---------------------------------------------------------------------------
extern "C" void kernel_launch(void* const* d_in, const int* in_sizes, int n_in,
                              void* d_out, int out_size, void* d_ws, size_t ws_size,
                              hipStream_t stream)
{
    // Input layout: rays + pytree leaves of params (alphabetical dict keys:
    // alpha{b,w}, feat{b,w}, pts[0..7]{b,w}, rgb{b,w}, views{b,w}).
    int raysIdx, pbase;
    if (in_sizes[0] == NRAYS * 6) { raysIdx = 0; pbase = 1; }
    else                          { raysIdx = n_in - 1; pbase = 0; }
    const float* rays    = (const float*)d_in[raysIdx];
    const float* alpha_b = (const float*)d_in[pbase + 0];
    const float* alpha_w = (const float*)d_in[pbase + 1];
    const float* feat_b  = (const float*)d_in[pbase + 2];
    const float* feat_w  = (const float*)d_in[pbase + 3];
    const float* pts_b[8]; const float* pts_w[8];
    for (int i = 0; i < 8; ++i) {
        pts_b[i] = (const float*)d_in[pbase + 4 + 2 * i];
        pts_w[i] = (const float*)d_in[pbase + 5 + 2 * i];
    }
    const float* rgb_b   = (const float*)d_in[pbase + 20];
    const float* rgb_w   = (const float*)d_in[pbase + 21];
    const float* views_b = (const float*)d_in[pbase + 22];
    const float* views_w = (const float*)d_in[pbase + 23];

    // workspace carve-out
    char* wsp = (char*)d_ws;
    size_t off = 0;
    auto carve = [&](size_t bytes) -> char* {
        char* p = wsp + off;
        off += (bytes + 255) & ~(size_t)255;
        return p;
    };
    // packed f16 weights (offsets in halfs)
    const size_t oW0 = 0, oW1 = 16384, oW2 = 81920, oW3 = 147456, oW4 = 212992;
    const size_t oW5 = 278528, oW6 = 360448, oW7 = 425984;
    const size_t oAl = 491520, oFt = 495616, oVw = 561152, oRg = 598016, oEnd = 600064;
    _Float16* WP   = (_Float16*)carve(oEnd * 2);
    _Float16* X0   = (_Float16*)carve((size_t)MF * 64 * 2);
    _Float16* Abuf = (_Float16*)carve((size_t)MF * 256 * 2);
    _Float16* Bbuf = (_Float16*)carve((size_t)MF * 256 * 2);
    float*    RAW  = (float*)   carve((size_t)MF * 4 * 4);
    _Float16* DENC = (_Float16*)carve((size_t)NRAYS * 32 * 2);
    float*    SCALE= (float*)   carve((size_t)NRAYS * 4);
    float*    ZC   = (float*)   carve((size_t)MC * 4);
    float*    W0   = (float*)   carve((size_t)MC * 4);
    float*    ZS   = (float*)   carve((size_t)MC * 4);
    float*    ZALL = (float*)   carve((size_t)MF * 4);
    (void)ws_size; (void)out_size;
    float* OUT = (float*)d_out;           // rgb[6144] depth[2048] acc[2048] weights[262144]

    // ---- pack all weights into WMMA B-fragment order ----
    auto pack = [&](const float* W, _Float16* dst, int K, int N, int Kp, int Np) {
        int tot = (Kp / 32) * (Np / 16) * 32;
        pack_weight<<<(tot + 255) / 256, 256, 0, stream>>>(W, dst, K, N, Kp, Np);
    };
    pack(pts_w[0], WP + oW0,  63, 256,  64, 256);
    pack(pts_w[1], WP + oW1, 256, 256, 256, 256);
    pack(pts_w[2], WP + oW2, 256, 256, 256, 256);
    pack(pts_w[3], WP + oW3, 256, 256, 256, 256);
    pack(pts_w[4], WP + oW4, 256, 256, 256, 256);
    pack(pts_w[5], WP + oW5, 319, 256, 320, 256);
    pack(pts_w[6], WP + oW6, 256, 256, 256, 256);
    pack(pts_w[7], WP + oW7, 256, 256, 256, 256);
    pack(alpha_w,  WP + oAl, 256,   1, 256,  16);
    pack(feat_w,   WP + oFt, 256, 256, 256, 256);
    pack(views_w,  WP + oVw, 283, 128, 288, 128);
    pack(rgb_w,    WP + oRg, 128,   3, 128,  16);

    const dim3 blk(128);
    encode_dirs<<<NRAYS / 256, 256, 0, stream>>>(rays, DENC, SCALE);

    // ================= coarse pass (sigma only) =================
    gen_z_coarse<<<MC / 256, 256, 0, stream>>>(ZC);
    encode_pts<<<MC / 256, 256, 0, stream>>>(rays, ZC, X0, NSC, MC);

    gemm_wmma<0,4,0, 2,16, 64,  0,256><<<dim3(4, MC/64), blk, 0, stream>>>(X0,   nullptr, WP+oW0, pts_b[0], Abuf, 0);
    gemm_wmma<0,4,0, 8,16,256,  0,256><<<dim3(4, MC/64), blk, 0, stream>>>(Abuf, nullptr, WP+oW1, pts_b[1], Bbuf, 0);
    gemm_wmma<0,4,0, 8,16,256,  0,256><<<dim3(4, MC/64), blk, 0, stream>>>(Bbuf, nullptr, WP+oW2, pts_b[2], Abuf, 0);
    gemm_wmma<0,4,0, 8,16,256,  0,256><<<dim3(4, MC/64), blk, 0, stream>>>(Abuf, nullptr, WP+oW3, pts_b[3], Bbuf, 0);
    gemm_wmma<0,4,0, 8,16,256,  0,256><<<dim3(4, MC/64), blk, 0, stream>>>(Bbuf, nullptr, WP+oW4, pts_b[4], Abuf, 0);
    gemm_wmma<1,4,0,10,16,256, 64,256><<<dim3(4, MC/64), blk, 0, stream>>>(Abuf, X0,      WP+oW5, pts_b[5], Bbuf, 0);
    gemm_wmma<0,4,0, 8,16,256,  0,256><<<dim3(4, MC/64), blk, 0, stream>>>(Bbuf, nullptr, WP+oW6, pts_b[6], Abuf, 0);
    gemm_wmma<0,4,0, 8,16,256,  0,256><<<dim3(4, MC/64), blk, 0, stream>>>(Abuf, nullptr, WP+oW7, pts_b[7], Bbuf, 0);
    gemm_wmma<0,1,2, 8, 1,256,  0,  0><<<dim3(1, MC/64), blk, 0, stream>>>(Bbuf, nullptr, WP+oAl, alpha_b,  RAW,  0);

    render_rays<<<NRAYS / 256, 256, 0, stream>>>(RAW, ZC, SCALE, NSC, W0, nullptr, nullptr, nullptr);
    sample_pdf_k<<<NRAYS / 256, 256, 0, stream>>>(W0, ZC, ZS);
    sort_z<<<NRAYS / 256, 256, 0, stream>>>(ZC, ZS, ZALL);

    // ================= fine pass (full network) =================
    encode_pts<<<MF / 256, 256, 0, stream>>>(rays, ZALL, X0, NSF, MF);

    gemm_wmma<0,4,0, 2,16, 64,  0,256><<<dim3(4, MF/64), blk, 0, stream>>>(X0,   nullptr, WP+oW0, pts_b[0], Abuf, 0);
    gemm_wmma<0,4,0, 8,16,256,  0,256><<<dim3(4, MF/64), blk, 0, stream>>>(Abuf, nullptr, WP+oW1, pts_b[1], Bbuf, 0);
    gemm_wmma<0,4,0, 8,16,256,  0,256><<<dim3(4, MF/64), blk, 0, stream>>>(Bbuf, nullptr, WP+oW2, pts_b[2], Abuf, 0);
    gemm_wmma<0,4,0, 8,16,256,  0,256><<<dim3(4, MF/64), blk, 0, stream>>>(Abuf, nullptr, WP+oW3, pts_b[3], Bbuf, 0);
    gemm_wmma<0,4,0, 8,16,256,  0,256><<<dim3(4, MF/64), blk, 0, stream>>>(Bbuf, nullptr, WP+oW4, pts_b[4], Abuf, 0);
    gemm_wmma<1,4,0,10,16,256, 64,256><<<dim3(4, MF/64), blk, 0, stream>>>(Abuf, X0,      WP+oW5, pts_b[5], Bbuf, 0);
    gemm_wmma<0,4,0, 8,16,256,  0,256><<<dim3(4, MF/64), blk, 0, stream>>>(Bbuf, nullptr, WP+oW6, pts_b[6], Abuf, 0);
    gemm_wmma<0,4,0, 8,16,256,  0,256><<<dim3(4, MF/64), blk, 0, stream>>>(Abuf, nullptr, WP+oW7, pts_b[7], Bbuf, 0);
    // heads: sigma, feat (no relu), views (concat feat + dir-enc), rgb
    gemm_wmma<0,1,2, 8, 1,256,  0,  0><<<dim3(1, MF/64), blk, 0, stream>>>(Bbuf, nullptr, WP+oAl, alpha_b,  RAW,  0);
    gemm_wmma<0,4,1, 8,16,256,  0,256><<<dim3(4, MF/64), blk, 0, stream>>>(Bbuf, nullptr, WP+oFt, feat_b,   Abuf, 0);
    gemm_wmma<2,4,0, 9, 8,256, 32,128><<<dim3(2, MF/64), blk, 0, stream>>>(Abuf, DENC,    WP+oVw, views_b,  Bbuf, NSF);
    gemm_wmma<0,1,3, 4, 1,128,  0,  0><<<dim3(1, MF/64), blk, 0, stream>>>(Bbuf, nullptr, WP+oRg, rgb_b,    RAW,  0);

    render_rays<<<NRAYS / 256, 256, 0, stream>>>(RAW, ZALL, SCALE, NSF,
                                                 OUT + 10240,      // weights
                                                 OUT,              // rgb_map
                                                 OUT + 6144,       // depth_map
                                                 OUT + 8192);      // acc_map
}
